// CnnReservoirPolicy_2302102470818
// MI455X (gfx1250) — matrix operand order repaired and checked by reading, not compiled, not run on hardware
//
#include <hip/hip_runtime.h>
#include <hip/hip_bf16.h>
#include <math.h>

// ---------------- CDNA5 WMMA types ----------------
typedef _Float16 v16h __attribute__((ext_vector_type(16)));
typedef _Float16 v8h  __attribute__((ext_vector_type(8)));
typedef float    v8f  __attribute__((ext_vector_type(8)));

#define LDSL 40   // LDS row stride in halves (80B, 16B aligned)

// A fragment (16x32 f16): lanes 0-15 row M=lane, K={0..7,16..23}; lanes 16-31 row M=lane-16, K={8..15,24..31}
__device__ __forceinline__ v16h frag_A(const _Float16* lds, int rowBase, int lane) {
  int r  = lane & 15;
  int hi = lane >> 4;
  const _Float16* p = lds + (rowBase + r) * LDSL;
  v8h a0 = *(const v8h*)(p + hi * 8);
  v8h a1 = *(const v8h*)(p + 16 + hi * 8);
  v16h out;
#pragma unroll
  for (int i = 0; i < 8; ++i) { out[i] = a0[i]; out[i + 8] = a1[i]; }
  return out;
}

// B fragment (32x16 f16): lanes 0-15 col N=lane K={0..7,16..23}; lanes 16-31 col N=lane-16 K={8..15,24..31}
__device__ __forceinline__ v16h frag_B(const _Float16* lds, int colBase, int lane) {
  int c  = lane & 15;
  int hi = lane >> 4;
  const _Float16* p = lds + (colBase + c) * LDSL;
  v8h b0 = *(const v8h*)(p + hi * 8);
  v8h b1 = *(const v8h*)(p + 16 + hi * 8);
  v16h out;
#pragma unroll
  for (int i = 0; i < 8; ++i) { out[i] = b0[i]; out[i + 8] = b1[i]; }
  return out;
}

// ---------------- layout prep kernels ----------------
__global__ void k_nchw2nhwc(const float* __restrict__ in, _Float16* __restrict__ out,
                            int NB, int C, int H, int W) {
  size_t i = (size_t)blockIdx.x * blockDim.x + threadIdx.x;
  size_t total = (size_t)NB * C * H * W;
  if (i >= total) return;
  int c = (int)(i % C); size_t r = i / C;
  int w = (int)(r % W); r /= W;
  int h = (int)(r % H); r /= H;
  int n = (int)r;
  out[i] = (_Float16)in[(((size_t)n * C + c) * H + h) * W + w];
}

// OIHW f32 -> [Cout][Kpad] f16, K ordered (kh, kw, ci), zero-padded to Kpad
__global__ void k_wreorder(const float* __restrict__ w, _Float16* __restrict__ out,
                           int Cout, int Cin, int KH, int KW, int Kpad) {
  int i = blockIdx.x * blockDim.x + threadIdx.x;
  if (i >= Cout * Kpad) return;
  int k = i % Kpad, o = i / Kpad;
  int K = Cin * KH * KW;
  _Float16 v = (_Float16)0.f;
  if (k < K) {
    int ci = k % Cin; int r = k / Cin;
    int kw = r % KW;  int kh = r / KW;
    v = (_Float16)w[(((size_t)o * Cin + ci) * KH + kh) * KW + kw];
  }
  out[i] = v;
}

// ---------------- implicit-GEMM conv (WMMA f16 -> f32 acc) ----------------
// Workgroup: 128 threads (4 waves), tile BM=64 x BN=64, BK=32, wave = 32x32 (2x2 WMMA).
// B (weight) tile staged memory->LDS with GLOBAL_LOAD_ASYNC_TO_LDS_B128 (ASYNCcnt path).
template <int KH, int KW, int S, int P>
__global__ __launch_bounds__(128) void k_conv_wmma(
    const _Float16* __restrict__ x, const _Float16* __restrict__ wB,
    _Float16* __restrict__ y,
    int H, int W, int Ho, int Wo, int Cin, int Cout, int Ktot, int Kpad) {
  __shared__ __align__(16) _Float16 sA[64 * LDSL];
  __shared__ __align__(16) _Float16 sB[64 * LDSL];

  const int t    = threadIdx.x;
  const int lane = t & 31;
  const int wid  = t >> 5;
  const int wm   = wid >> 1;   // 0..1
  const int wn   = wid & 1;    // 0..1

  const int mTile = blockIdx.x * 64;
  const int nTile = blockIdx.y * 64;

  // cooperative loader coords: each thread covers 16 contiguous k of one tile row
  const int lrow = t >> 1;            // 0..63
  const int lk16 = (t & 1) * 16;      // 0 or 16

  // decode output pixel of the A row this thread stages
  const int am   = mTile + lrow;
  const int HoWo = Ho * Wo;
  const int an   = am / HoWo;
  int rem        = am - an * HoWo;
  const int aho  = rem / Wo;
  const int awo  = rem - aho * Wo;
  const int hb   = aho * S - P;
  const int wb   = awo * S - P;
  const _Float16* xbase = x + (size_t)an * H * W * Cin;
  __builtin_prefetch(xbase, 0, 1);                       // global_prefetch_b8

  // async B staging setup: this thread owns 2 x 16B chunks of row lrow
  const unsigned long long wb64 = (unsigned long long)(uintptr_t)wB;
  const unsigned ldsB0 = (unsigned)(uintptr_t)(&sB[lrow * LDSL + lk16]);        // LDS byte offset
  const unsigned ldsB1 = ldsB0 + 16u;
  const unsigned gB0   = (unsigned)(((nTile + lrow) * Kpad + lk16) * 2);        // global byte offset

  v8f vzero = {};
  v8f acc[2][2];
#pragma unroll
  for (int i = 0; i < 2; ++i)
#pragma unroll
    for (int j = 0; j < 2; ++j) acc[i][j] = vzero;

  for (int kk = 0; kk < Kpad; kk += 32) {
    __syncthreads();        // previous-iteration consumers done before restaging
    // ---- async stage B tile (weights, zero-padded to Kpad -> no guard) ----
    {
      unsigned vo = gB0 + (unsigned)(kk * 2);
      asm volatile("global_load_async_to_lds_b128 %0, %1, %2"
                   :: "v"(ldsB0), "v"(vo), "s"(wb64) : "memory");
      asm volatile("global_load_async_to_lds_b128 %0, %1, %2"
                   :: "v"(ldsB1), "v"(vo + 16u), "s"(wb64) : "memory");
    }
    // ---- stage A (guarded im2col gather, f16) ----
    {
      int kstart = kk + lk16;
      int t1 = kstart / Cin;
      int ci = kstart - t1 * Cin;
      int kh = t1 / KW;
      int kw = t1 - kh * KW;
#pragma unroll 4
      for (int j = 0; j < 16; ++j) {
        float v = 0.f;
        if (kstart + j < Ktot) {
          int hin = hb + kh, win = wb + kw;
          if (hin >= 0 && hin < H && win >= 0 && win < W)
            v = (float)xbase[((size_t)hin * W + win) * Cin + ci];
        }
        sA[lrow * LDSL + lk16 + j] = (_Float16)v;
        if (++ci == Cin) { ci = 0; if (++kw == KW) { kw = 0; ++kh; } }
      }
    }
    asm volatile("s_wait_asynccnt 0x0" ::: "memory");     // our async B chunks landed in LDS
    __syncthreads();                                      // all waves' staging visible
    // ---- WMMA 2x2 ----
    v16h afr[2], bfr[2];
#pragma unroll
    for (int i = 0; i < 2; ++i) afr[i] = frag_A(sA, wm * 32 + i * 16, lane);
#pragma unroll
    for (int j = 0; j < 2; ++j) bfr[j] = frag_B(sB, wn * 32 + j * 16, lane);
#pragma unroll
    for (int i = 0; i < 2; ++i)
#pragma unroll
      for (int j = 0; j < 2; ++j)
        acc[i][j] = __builtin_amdgcn_wmma_f32_16x16x32_f16(
            false, afr[i], false, bfr[j], (short)0, acc[i][j], false, false);
  }

  // ---- store D (C/D layout: lane -> N, vgpr -> M) ----
  const int cn = lane & 15;
  const int rb = (lane >> 4) * 8;
#pragma unroll
  for (int i = 0; i < 2; ++i)
#pragma unroll
    for (int j = 0; j < 2; ++j) {
      int col  = nTile + wn * 32 + j * 16 + cn;
      int row0 = mTile + wm * 32 + i * 16 + rb;
#pragma unroll
      for (int v = 0; v < 8; ++v)
        y[(size_t)(row0 + v) * Cout + col] = (_Float16)acc[i][j][v];
    }
}

// ---------------- BatchNorm (training stats) ----------------
__global__ void k_bn_stats(const _Float16* __restrict__ y, float* __restrict__ stats,
                           int M, int C) {
  __shared__ float ss[256], sq[256];
  int c = blockIdx.x;
  float s = 0.f, q = 0.f;
  for (int m = blockIdx.y * blockDim.x + threadIdx.x; m < M; m += gridDim.y * blockDim.x) {
    float v = (float)y[(size_t)m * C + c];
    s += v; q += v * v;
  }
  ss[threadIdx.x] = s; sq[threadIdx.x] = q;
  __syncthreads();
  for (int o = 128; o > 0; o >>= 1) {
    if (threadIdx.x < o) { ss[threadIdx.x] += ss[threadIdx.x + o]; sq[threadIdx.x] += sq[threadIdx.x + o]; }
    __syncthreads();
  }
  if (threadIdx.x == 0) { atomicAdd(&stats[c], ss[0]); atomicAdd(&stats[C + c], sq[0]); }
}

__global__ void k_bn_apply(_Float16* __restrict__ y, const float* __restrict__ stats,
                           const float* __restrict__ g, const float* __restrict__ be,
                           int M, int C) {
  size_t i = (size_t)blockIdx.x * blockDim.x + threadIdx.x;
  size_t total = (size_t)M * C;
  if (i >= total) return;
  int c = (int)(i % C);
  float inv_m = 1.0f / (float)M;
  float mean = stats[c] * inv_m;
  float var  = stats[C + c] * inv_m - mean * mean;
  float sc   = g[c] * rsqrtf(var + 1e-5f);
  float v    = ((float)y[i] - mean) * sc + be[c];
  y[i] = (_Float16)fmaxf(v, 0.0f);
}

// ---------------- maxpool 3x3 s2 p1 (NHWC f16) ----------------
__global__ void k_maxpool(const _Float16* __restrict__ in, _Float16* __restrict__ out,
                          int NB, int C, int H, int W, int Ho, int Wo) {
  size_t i = (size_t)blockIdx.x * blockDim.x + threadIdx.x;
  size_t total = (size_t)NB * Ho * Wo * C;
  if (i >= total) return;
  int c = (int)(i % C); size_t r = i / C;
  int wo = (int)(r % Wo); r /= Wo;
  int ho = (int)(r % Ho); r /= Ho;
  int n  = (int)r;
  float best = -INFINITY;
  for (int dh = 0; dh < 3; ++dh) {
    int h = ho * 2 - 1 + dh;
    if (h < 0 || h >= H) continue;
    for (int dw = 0; dw < 3; ++dw) {
      int w = wo * 2 - 1 + dw;
      if (w < 0 || w >= W) continue;
      float v = (float)in[(((size_t)n * H + h) * W + w) * C + c];
      best = fmaxf(best, v);
    }
  }
  out[i] = (_Float16)best;
}

// ---------------- tail: features, reservoir, head (f32, memory-bound) ----------------
__global__ void k_feat(const _Float16* __restrict__ y4, float* __restrict__ feat) {
  int i = blockIdx.x * blockDim.x + threadIdx.x;           // 8*16*512
  if (i >= 8 * 16 * 512) return;
  int c = i % 512; int bt = i / 512;
  int tt = bt % 16; int b = bt / 16;
  float s = 0.f;
  for (int cam = 0; cam < 2; ++cam) {
    size_t base = (size_t)((b * 2 + cam) * 16 + tt) * 16 * 512;
    for (int sp = 0; sp < 16; ++sp) s += (float)y4[base + sp * 512 + c];
  }
  feat[i] = s * (1.0f / 32.0f);
}

__global__ void k_stepin(const float* __restrict__ feat, const float* __restrict__ state,
                         const float* __restrict__ sW, const float* __restrict__ sb,
                         float* __restrict__ si) {
  int i = blockIdx.x * blockDim.x + threadIdx.x;           // 128*576
  if (i >= 128 * 576) return;
  int f = i % 576; int bt = i / 576;
  float v;
  if (f < 512) v = feat[bt * 512 + f];
  else {
    int j = f - 512;
    float a = sb[j];
    for (int d = 0; d < 14; ++d) a += state[bt * 14 + d] * sW[j * 14 + d];
    v = fmaxf(a, 0.0f);
  }
  si[i] = v;
}

__global__ void k_xin(const float* __restrict__ si, const float* __restrict__ Win,
                      const float* __restrict__ brs, float* __restrict__ xin) {
  int i = blockIdx.x * blockDim.x + threadIdx.x;           // 128*1024
  if (i >= 128 * 1024) return;
  int n = i % 1024; int bt = i / 1024;
  float a = brs[n];
  const float* s = si + bt * 576;
  const float* w = Win + (size_t)n * 576;
  for (int d = 0; d < 576; ++d) a += s[d] * w[d];
  xin[i] = a;
}

__global__ void k_res(const float* __restrict__ xin, const float* __restrict__ rin,
                      const float* __restrict__ Wres, float* __restrict__ rout,
                      float* __restrict__ rseq, int t) {
  int i = blockIdx.x * blockDim.x + threadIdx.x;           // 8*1024
  if (i >= 8 * 1024) return;
  int n = i % 1024; int b = i / 1024;
  float a = xin[(b * 16 + t) * 1024 + n];
  const float* r = rin + b * 1024;
  const float* w = Wres + (size_t)n * 1024;
  for (int k = 0; k < 1024; ++k) a += r[k] * w[k];
  float v = 0.5f * rin[b * 1024 + n] + 0.5f * tanhf(a);
  rout[b * 1024 + n] = v;
  rseq[(b * 16 + t) * 1024 + n] = v;
}

__global__ void k_head1(const float* __restrict__ rseq, const float* __restrict__ hW1,
                        const float* __restrict__ hb1, float* __restrict__ h) {
  int i = blockIdx.x * blockDim.x + threadIdx.x;           // 8*512
  if (i >= 8 * 512) return;
  int j = i % 512; int b = i / 512;
  const float* r = rseq + b * 16384;
  const float* w = hW1 + (size_t)j * 16384;
  float a = hb1[j];
  for (int k = 0; k < 16384; ++k) a += r[k] * w[k];
  h[i] = fmaxf(a, 0.0f);
}

__global__ void k_head2(const float* __restrict__ h, const float* __restrict__ hW2,
                        const float* __restrict__ hb2, float* __restrict__ out) {
  int i = blockIdx.x * blockDim.x + threadIdx.x;           // 8*112
  if (i >= 8 * 112) return;
  int o = i % 112; int b = i / 112;
  const float* hh = h + b * 512;
  const float* w  = hW2 + (size_t)o * 512;
  float a = hb2[o];
  for (int k = 0; k < 512; ++k) a += hh[k] * w[k];
  out[i] = a;
}

// ---------------- host orchestration ----------------
extern "C" void kernel_launch(void* const* d_in, const int* in_sizes, int n_in,
                              void* d_out, int out_size, void* d_ws, size_t ws_size,
                              hipStream_t stream) {
  (void)in_sizes; (void)n_in; (void)out_size; (void)ws_size;
  const float* state  = (const float*)d_in[0];
  const float* images = (const float*)d_in[1];
  const float* w1 = (const float*)d_in[2];
  const float* g1 = (const float*)d_in[3];
  const float* be1 = (const float*)d_in[4];
  const float* w2 = (const float*)d_in[5];
  const float* g2 = (const float*)d_in[6];
  const float* be2 = (const float*)d_in[7];
  const float* w3 = (const float*)d_in[8];
  const float* g3 = (const float*)d_in[9];
  const float* be3 = (const float*)d_in[10];
  const float* w4 = (const float*)d_in[11];
  const float* g4 = (const float*)d_in[12];
  const float* be4 = (const float*)d_in[13];
  const float* sWp = (const float*)d_in[14];
  const float* sbp = (const float*)d_in[15];
  const float* Win = (const float*)d_in[16];
  const float* Wres = (const float*)d_in[17];
  const float* brs = (const float*)d_in[18];
  const float* hW1 = (const float*)d_in[19];
  const float* hb1 = (const float*)d_in[20];
  const float* hW2 = (const float*)d_in[21];
  const float* hb2 = (const float*)d_in[22];
  float* out = (float*)d_out;

  size_t off = 0;
  auto alloc = [&](size_t bytes) -> void* {
    off = (off + 255) & ~(size_t)255;
    void* p = (char*)d_ws + off;
    off += bytes;
    return p;
  };

  const int NB = 256;                                    // 8 * 2 * 16 frames
  const int K1 = 147, K1p = 160;                         // conv1 K padded to BK multiple
  const int K2 = 576, K2p = 576;
  const int K3 = 1152, K3p = 1152;
  const int K4 = 2304, K4p = 2304;

  _Float16* x0  = (_Float16*)alloc((size_t)NB * 112 * 112 * 3 * 2);
  _Float16* w1r = (_Float16*)alloc((size_t)64 * K1p * 2);
  _Float16* w2r = (_Float16*)alloc((size_t)128 * K2p * 2);
  _Float16* w3r = (_Float16*)alloc((size_t)256 * K3p * 2);
  _Float16* w4r = (_Float16*)alloc((size_t)512 * K4p * 2);
  _Float16* y1  = (_Float16*)alloc((size_t)NB * 56 * 56 * 64 * 2);
  _Float16* p1  = (_Float16*)alloc((size_t)NB * 28 * 28 * 64 * 2);
  _Float16* y2  = (_Float16*)alloc((size_t)NB * 14 * 14 * 128 * 2);
  _Float16* y3  = (_Float16*)alloc((size_t)NB * 7 * 7 * 256 * 2);
  _Float16* y4  = (_Float16*)alloc((size_t)NB * 4 * 4 * 512 * 2);
  float* stats = (float*)alloc(2 * 512 * sizeof(float));
  float* feat  = (float*)alloc(8 * 16 * 512 * sizeof(float));
  float* si    = (float*)alloc(128 * 576 * sizeof(float));
  float* xin   = (float*)alloc(128 * 1024 * sizeof(float));
  float* rA    = (float*)alloc(8 * 1024 * sizeof(float));
  float* rB    = (float*)alloc(8 * 1024 * sizeof(float));
  float* rseq  = (float*)alloc(8 * 16 * 1024 * sizeof(float));
  float* h     = (float*)alloc(8 * 512 * sizeof(float));

  auto blocks = [](size_t n) { return (unsigned)((n + 255) / 256); };

  // layout prep
  k_nchw2nhwc<<<blocks((size_t)NB * 3 * 112 * 112), 256, 0, stream>>>(images, x0, NB, 3, 112, 112);
  k_wreorder<<<blocks(64 * K1p), 256, 0, stream>>>(w1, w1r, 64, 3, 7, 7, K1p);
  k_wreorder<<<blocks(128 * K2p), 256, 0, stream>>>(w2, w2r, 128, 64, 3, 3, K2p);
  k_wreorder<<<blocks(256 * K3p), 256, 0, stream>>>(w3, w3r, 256, 128, 3, 3, K3p);
  k_wreorder<<<blocks(512 * K4p), 256, 0, stream>>>(w4, w4r, 512, 256, 3, 3, K4p);

  // conv1 + BN + ReLU + maxpool
  k_conv_wmma<7, 7, 2, 3><<<dim3(12544, 1), 128, 0, stream>>>(x0, w1r, y1, 112, 112, 56, 56, 3, 64, K1, K1p);
  hipMemsetAsync(stats, 0, 2 * 64 * sizeof(float), stream);
  k_bn_stats<<<dim3(64, 64), 256, 0, stream>>>(y1, stats, NB * 56 * 56, 64);
  k_bn_apply<<<blocks((size_t)NB * 56 * 56 * 64), 256, 0, stream>>>(y1, stats, g1, be1, NB * 56 * 56, 64);
  k_maxpool<<<blocks((size_t)NB * 28 * 28 * 64), 256, 0, stream>>>(y1, p1, NB, 64, 56, 56, 28, 28);

  // conv2 + BN + ReLU
  k_conv_wmma<3, 3, 2, 1><<<dim3(784, 2), 128, 0, stream>>>(p1, w2r, y2, 28, 28, 14, 14, 64, 128, K2, K2p);
  hipMemsetAsync(stats, 0, 2 * 128 * sizeof(float), stream);
  k_bn_stats<<<dim3(128, 32), 256, 0, stream>>>(y2, stats, NB * 14 * 14, 128);
  k_bn_apply<<<blocks((size_t)NB * 14 * 14 * 128), 256, 0, stream>>>(y2, stats, g2, be2, NB * 14 * 14, 128);

  // conv3 + BN + ReLU
  k_conv_wmma<3, 3, 2, 1><<<dim3(196, 4), 128, 0, stream>>>(y2, w3r, y3, 14, 14, 7, 7, 128, 256, K3, K3p);
  hipMemsetAsync(stats, 0, 2 * 256 * sizeof(float), stream);
  k_bn_stats<<<dim3(256, 16), 256, 0, stream>>>(y3, stats, NB * 7 * 7, 256);
  k_bn_apply<<<blocks((size_t)NB * 7 * 7 * 256), 256, 0, stream>>>(y3, stats, g3, be3, NB * 7 * 7, 256);

  // conv4 + BN + ReLU
  k_conv_wmma<3, 3, 2, 1><<<dim3(64, 8), 128, 0, stream>>>(y3, w4r, y4, 7, 7, 4, 4, 256, 512, K4, K4p);
  hipMemsetAsync(stats, 0, 2 * 512 * sizeof(float), stream);
  k_bn_stats<<<dim3(512, 8), 256, 0, stream>>>(y4, stats, NB * 4 * 4, 512);
  k_bn_apply<<<blocks((size_t)NB * 4 * 4 * 512), 256, 0, stream>>>(y4, stats, g4, be4, NB * 4 * 4, 512);

  // features + state embed + input drive
  k_feat<<<blocks(8 * 16 * 512), 256, 0, stream>>>(y4, feat);
  k_stepin<<<blocks(128 * 576), 256, 0, stream>>>(feat, state, sWp, sbp, si);
  k_xin<<<blocks(128 * 1024), 256, 0, stream>>>(si, Win, brs, xin);

  // reservoir scan (sequential dependence -> 16 launches)
  hipMemsetAsync(rA, 0, 8 * 1024 * sizeof(float), stream);
  float* rcur = rA; float* rnext = rB;
  for (int t = 0; t < 16; ++t) {
    k_res<<<blocks(8 * 1024), 256, 0, stream>>>(xin, rcur, Wres, rnext, rseq, t);
    float* tmp = rcur; rcur = rnext; rnext = tmp;
  }

  // readout head
  k_head1<<<blocks(8 * 512), 256, 0, stream>>>(rseq, hW1, hb1, h);
  k_head2<<<blocks(8 * 112), 256, 0, stream>>>(h, hW2, hb2, out);
}
